// Encoder_75703093559814
// MI455X (gfx1250) — matrix operand order repaired and checked by reading, not compiled, not run on hardware
//
#include <hip/hip_runtime.h>
#include <hip/hip_bf16.h>

// ---------------------------------------------------------------------------
// MI455X (gfx1250) fused transformer encoder layer, round 3.
//  * all GEMMs on v_wmma_f32_16x16x32_f16 (f16 in, f32 accumulate)
//  * global->LDS via global_load_async_to_lds_b128 (ASYNCcnt); GEMM runs a
//    depth-2 pipeline over 3 LDS buffers using in-order partial waits
//    (s_wait_asynccnt 4 retires exactly the oldest tile)
//  * weights pre-transposed to [N][K] so every B-fragment is two 8-half
//    LDS reads (k 8h.., 16+8h..) of one row, matching the ISA B layout
//  * flash attention: V written pre-transposed by the QKV epilogue
// ---------------------------------------------------------------------------

typedef _Float16 half8  __attribute__((ext_vector_type(8)));
typedef _Float16 half16 __attribute__((ext_vector_type(16)));
typedef float    float8 __attribute__((ext_vector_type(8)));

#define MDIM   1024
#define FFNH   2048
#define NHEAD  16
#define HDIM   64
#define BATCH  2
#define SEQ    2048
#define ROWS   (BATCH * SEQ)      // 4096 token rows

// ---- helpers --------------------------------------------------------------

__device__ inline half16 cat8(half8 lo, half8 hi) {
  return __builtin_shufflevector(lo, hi, 0,1,2,3,4,5,6,7,8,9,10,11,12,13,14,15);
}

__device__ inline float8 wmma16(half16 a, half16 b, float8 c) {
  return __builtin_amdgcn_wmma_f32_16x16x32_f16(false, a, false, b, (short)0, c,
                                                false, false);
}

// Async DMA of 16 bytes global -> LDS (per lane). Tracked by ASYNCcnt.
// The LDS aperture occupies addr[63:32], so the low 32 bits of a generic
// pointer to __shared__ are exactly the LDS byte address the VDST wants.
__device__ inline void async_b128(const void* gbase, uint32_t byte_off,
                                  void* lds_ptr) {
  uint32_t lds32 = (uint32_t)(uintptr_t)lds_ptr;
  asm volatile("global_load_async_to_lds_b128 %0, %1, %2"
               :: "v"(lds32), "v"(byte_off),
                  "s"((unsigned long long)(uintptr_t)gbase)
               : "memory");
}
// Async completions are in-order, so waiting ASYNCcnt<=N retires exactly the
// oldest (cnt-N) operations: partial waits drive the depth-2 pipeline.
template <int N>
__device__ inline void wait_async() {
  asm volatile("s_wait_asynccnt %0" :: "n"(N) : "memory");
}

// ---- cast / pack kernels --------------------------------------------------

__global__ void cast_f32_to_f16(const float* __restrict__ src,
                                _Float16* __restrict__ dst, int n) {
  int i = blockIdx.x * blockDim.x + threadIdx.x;
  if (i < n) dst[i] = (_Float16)src[i];
}

// src[R][C] f32 -> dst[C][R] f16  (weights become [N][K] for B-fragments)
__global__ void tcast_f32_to_f16(const float* __restrict__ src,
                                 _Float16* __restrict__ dst, int R, int C) {
  int i = blockIdx.x * blockDim.x + threadIdx.x;
  if (i >= R * C) return;
  int r = i / C, c = i % C;
  dst[(size_t)c * R + r] = (_Float16)src[i];
}

// Wq/Wk/Wv [H, M, D] -> wqkvT [3*M][M] with out-row = sel*1024 + h*64 + d
__global__ void pack_qkv_wT(const float* __restrict__ Wq,
                            const float* __restrict__ Wk,
                            const float* __restrict__ Wv,
                            _Float16* __restrict__ dst) {
  int i = blockIdx.x * blockDim.x + threadIdx.x;   // i < 3072*1024
  if (i >= 3 * MDIM * MDIM) return;
  int c = i >> 10;           // output row (N index)
  int m = i & 1023;          // K index
  int sel = c >> 10, cc = c & 1023;
  int h = cc >> 6, d = cc & 63;
  const float* W = (sel == 0) ? Wq : (sel == 1) ? Wk : Wv;
  dst[i] = (_Float16)W[((size_t)h * MDIM + m) * HDIM + d];
}

__global__ void pack_qkv_b(const float* __restrict__ bq,
                           const float* __restrict__ bk,
                           const float* __restrict__ bv,
                           float* __restrict__ dst) {
  int c = blockIdx.x * blockDim.x + threadIdx.x;
  if (c >= 3 * MDIM) return;
  int sel = c >> 10, cc = c & 1023;
  dst[c] = (sel == 0) ? bq[cc] : (sel == 1) ? bk[cc] : bv[cc];
}

// ---- generic WMMA GEMM ----------------------------------------------------
// C[M,N] = A[M,K] * BT[N,K]^T + bias[N]
// 128x128 block tile, BK=32, 256 threads = 8 wave32 waves (2x4), each wave a
// 64x32 tile (4x2 WMMA accumulators). Triple-buffered LDS fed by a depth-2
// async-DMA pipeline; one barrier per k-step.
// OUTMODE: 0 = f32 store, 1 = f16 store, 2 = QKV special (q,k row-major f16 +
// v transposed to vT[b,h,d,s]).

template <int OUTMODE, bool RELU>
__global__ __launch_bounds__(256)
void gemm_f16_wmma(const _Float16* __restrict__ A, const _Float16* __restrict__ BT,
                   const float* __restrict__ bias, void* __restrict__ Cout,
                   _Float16* __restrict__ vT, int Ndim, int Kdim) {
  __shared__ alignas(16) _Float16 As[3][128][40];   // [buf][m][k], 80B stride
  __shared__ alignas(16) _Float16 Bs[3][128][40];   // [buf][n][k]

  const int tid  = threadIdx.x;
  const int wid  = tid >> 5;
  const int lane = tid & 31;
  const int l    = lane & 15;
  const int hsel = lane >> 4;
  const int m0   = blockIdx.y * 128;
  const int n0   = blockIdx.x * 128;
  const int wm   = (wid & 1) * 64;
  const int wn   = (wid >> 1) * 32;

  // Per-thread DMA chunk geometry: 2 x 16B chunks per array per tile.
  int rC[2], kC[2];
  uint32_t aBase[2], bBase[2];
#pragma unroll
  for (int i = 0; i < 2; ++i) {
    int c = tid + i * 256;              // 512 chunks cover 128x32 halfs
    rC[i] = c >> 2;
    kC[i] = (c & 3) * 8;
    aBase[i] = (uint32_t)(((size_t)(m0 + rC[i]) * Kdim + kC[i]) * 2);
    bBase[i] = (uint32_t)(((size_t)(n0 + rC[i]) * Kdim + kC[i]) * 2);
  }
  auto issue_tile = [&](int it, int buf) {
    uint32_t kadd = (uint32_t)it * 64u;   // BK=32 halfs = 64 bytes
#pragma unroll
    for (int i = 0; i < 2; ++i) {
      async_b128(A,  aBase[i] + kadd, &As[buf][rC[i]][kC[i]]);
      async_b128(BT, bBase[i] + kadd, &Bs[buf][rC[i]][kC[i]]);
    }
  };

  float8 acc[4][2] = {};
  const int nIter = Kdim / 32;          // >= 32 for all our GEMMs
  issue_tile(0, 0);                     // depth-2 prologue: 8 async/wave
  issue_tile(1, 1);

  int p = 0;
  for (int it = 0; it < nIter; ++it) {
    // Retire the oldest tile (4 async/wave); keep the newer one in flight.
    if (it + 1 < nIter) wait_async<4>();
    else                wait_async<0>();
    __syncthreads();                    // tile `it` visible to all waves
    if (it + 2 < nIter) {
      int nb = p + 2; if (nb >= 3) nb -= 3;
      issue_tile(it + 2, nb);           // buffer last read in iter it-1
    }

    // Load ALL fragments first so one dscnt wait covers 8 chained WMMAs.
    half16 af[4], bf[2];
#pragma unroll
    for (int mt = 0; mt < 4; ++mt) {
      const _Float16* q = &As[p][wm + mt * 16 + l][hsel * 8];
      af[mt] = cat8(*(const half8*)q, *(const half8*)(q + 16));
    }
#pragma unroll
    for (int nt = 0; nt < 2; ++nt) {
      const _Float16* q = &Bs[p][wn + nt * 16 + l][hsel * 8];
      bf[nt] = cat8(*(const half8*)q, *(const half8*)(q + 16));
    }
#pragma unroll
    for (int nt = 0; nt < 2; ++nt)
#pragma unroll
      for (int mt = 0; mt < 4; ++mt)
        acc[mt][nt] = wmma16(af[mt], bf[nt], acc[mt][nt]);

    p = (p == 2) ? 0 : p + 1;
  }

  // Epilogue. C layout: VGPR r -> row r (lanes 0-15) / r+8 (lanes 16-31).
#pragma unroll
  for (int mt = 0; mt < 4; ++mt)
#pragma unroll
    for (int nt = 0; nt < 2; ++nt)
#pragma unroll
      for (int r = 0; r < 8; ++r) {
        int row = m0 + wm + mt * 16 + hsel * 8 + r;
        int col = n0 + wn + nt * 16 + l;
        float v = acc[mt][nt][r] + bias[col];
        if (RELU) v = fmaxf(v, 0.f);
        if (OUTMODE == 0) {
          ((float*)Cout)[(size_t)row * Ndim + col] = v;
        } else if (OUTMODE == 1) {
          ((_Float16*)Cout)[(size_t)row * Ndim + col] = (_Float16)v;
        } else {
          // QKV: cols [0,2048) = q|k row-major; [2048,3072) = v transposed.
          if (col < 2 * MDIM) {
            ((_Float16*)Cout)[(size_t)row * (2 * MDIM) + col] = (_Float16)v;
          } else {
            int cc = col - 2 * MDIM, hh = cc >> 6, d = cc & 63;
            int b = row >> 11, s = row & (SEQ - 1);
            vT[(((size_t)(b * NHEAD + hh) * HDIM) + d) * SEQ + s] = (_Float16)v;
          }
        }
      }
}

// ---- fused flash attention ------------------------------------------------
// Block = (64 q-rows, one head, one batch); 128 threads = 4 waves, each wave
// owns 16 q-rows. 64-key tiles streamed by async DMA (double-buffered; the
// long softmax+WMMA phase hides one tile of DMA latency), online softmax with
// __shfl_xor half-wave reductions, ctx held in WMMA f32 accumulators.

__global__ __launch_bounds__(128)
void attn_flash(const _Float16* __restrict__ qk,   // [ROWS][2048] f16 (q|k)
                const _Float16* __restrict__ vT,   // [B*H*D][SEQ]  f16
                const int* __restrict__ amask,     // [BATCH][SEQ]
                _Float16* __restrict__ ctx_out) {  // [ROWS][1024] f16
  const int qt = blockIdx.x, h = blockIdx.y, b = blockIdx.z;
  const int tid = threadIdx.x, wid = tid >> 5, lane = tid & 31;
  const int l = lane & 15, hsel = lane >> 4;
  const int qbase = qt * 64;

  __shared__ alignas(16) _Float16 Qs[64][72];        // [qrow][d]
  __shared__ alignas(16) _Float16 Ks[2][64][72];     // [buf][key][d]
  __shared__ alignas(16) _Float16 Vs[2][64][72];     // [buf][d][key]
  __shared__ alignas(16) _Float16 Ps[4][16][72];     // per-wave P staging

  // Async-load the Q tile once (512 chunks of 16B over 128 threads).
#pragma unroll
  for (int i = 0; i < 4; ++i) {
    int c = tid + i * 128;
    int row = c >> 3, dc = (c & 7) * 8;
    async_b128(qk,
        (uint32_t)(((size_t)(b * SEQ + qbase + row) * (2 * MDIM) + h * HDIM + dc) * 2),
        &Qs[row][dc]);
  }

  auto issue_kv = [&](int kt, int buf) {
    const int kbase = kt * 64;
#pragma unroll
    for (int i = 0; i < 4; ++i) {
      int c = tid + i * 128;
      int row = c >> 3, dc = (c & 7) * 8;   // row = key for K, = d for V
      async_b128(qk,
          (uint32_t)(((size_t)(b * SEQ + kbase + row) * (2 * MDIM) + MDIM + h * HDIM + dc) * 2),
          &Ks[buf][row][dc]);
      async_b128(vT,
          (uint32_t)(((size_t)((b * NHEAD + h) * HDIM + row) * SEQ + kbase + dc) * 2),
          &Vs[buf][row][dc]);
    }
  };

  float8 ctx[4] = {};
  float m[8], lsum[8];
#pragma unroll
  for (int r = 0; r < 8; ++r) { m[r] = -1e30f; lsum[r] = 0.f; }
  const float scale = 0.125f;   // 1/sqrt(64)

  half16 aq[2];
  int p = 0;
  issue_kv(0, 0);

  for (int kt = 0; kt < SEQ / 64; ++kt) {
    wait_async<0>();
    __syncthreads();
    if (kt == 0) {        // Qs ready after the first wait+barrier
      int row = wid * 16 + l;
#pragma unroll
      for (int kk = 0; kk < 2; ++kk) {
        const _Float16* q = &Qs[row][kk * 32 + hsel * 8];
        aq[kk] = cat8(*(const half8*)q, *(const half8*)(q + 16));
      }
    }
    if (kt + 1 < SEQ / 64) issue_kv(kt + 1, p ^ 1);
    const int kbase = kt * 64;

    // scores = q*k^T : per kk, load 4 B-fragments then 4 WMMAs back-to-back
    float8 sacc[4];
#pragma unroll
    for (int kk = 0; kk < 2; ++kk) {
      half16 bk[4];
#pragma unroll
      for (int nt = 0; nt < 4; ++nt) {
        const _Float16* q = &Ks[p][nt * 16 + l][kk * 32 + hsel * 8];
        bk[nt] = cat8(*(const half8*)q, *(const half8*)(q + 16));
      }
#pragma unroll
      for (int nt = 0; nt < 4; ++nt)
        sacc[nt] = (kk == 0) ? wmma16(aq[0], bk[nt], float8{})
                             : wmma16(aq[1], bk[nt], sacc[nt]);
    }

    float madd[4];
#pragma unroll
    for (int nt = 0; nt < 4; ++nt)
      madd[nt] = amask[b * SEQ + kbase + nt * 16 + l] ? 0.f : -1e9f;

    // Online softmax; row r lives in VGPR r across this half-wave's 16 lanes.
    float mnew[8];
#pragma unroll
    for (int r = 0; r < 8; ++r) {
      float t = -1e30f;
#pragma unroll
      for (int nt = 0; nt < 4; ++nt) {
        float sv = sacc[nt][r] * scale + madd[nt];
        sacc[nt][r] = sv;
        t = fmaxf(t, sv);
      }
#pragma unroll
      for (int off = 1; off < 16; off <<= 1)
        t = fmaxf(t, __shfl_xor(t, off, 32));
      mnew[r] = fmaxf(m[r], t);
    }
#pragma unroll
    for (int r = 0; r < 8; ++r) {
      float alpha = __expf(m[r] - mnew[r]);
      m[r] = mnew[r];
      float rs = 0.f;
#pragma unroll
      for (int nt = 0; nt < 4; ++nt) {
        float pv = __expf(sacc[nt][r] - mnew[r]);
        Ps[wid][r + hsel * 8][nt * 16 + l] = (_Float16)pv;
        rs += pv;
      }
#pragma unroll
      for (int off = 1; off < 16; off <<= 1)
        rs += __shfl_xor(rs, off, 32);
      lsum[r] = lsum[r] * alpha + rs;
#pragma unroll
      for (int dt = 0; dt < 4; ++dt) ctx[dt][r] *= alpha;
    }

    // DS ops are in-order per wave; drain + compiler fence, then re-read P
    // in A-fragment layout. Per-wave staging -> no block barrier needed.
    asm volatile("s_wait_dscnt 0" ::: "memory");

#pragma unroll
    for (int kk = 0; kk < 2; ++kk) {
      const _Float16* q = &Ps[wid][l][kk * 32 + hsel * 8];
      half16 ap = cat8(*(const half8*)q, *(const half8*)(q + 16));
      half16 bv[4];
#pragma unroll
      for (int dt = 0; dt < 4; ++dt) {
        const _Float16* v = &Vs[p][dt * 16 + l][kk * 32 + hsel * 8];
        bv[dt] = cat8(*(const half8*)v, *(const half8*)(v + 16));
      }
#pragma unroll
      for (int dt = 0; dt < 4; ++dt)
        ctx[dt] = wmma16(ap, bv[dt], ctx[dt]);
    }
    p ^= 1;
  }

  // Normalize, store ctx as [b, s, h*64+d] f16 (A operand of the Wo GEMM).
#pragma unroll
  for (int r = 0; r < 8; ++r) {
    float inv = 1.f / lsum[r];
    int row = qbase + wid * 16 + hsel * 8 + r;
#pragma unroll
    for (int dt = 0; dt < 4; ++dt) {
      int col = h * HDIM + dt * 16 + l;
      ctx_out[(size_t)(b * SEQ + row) * MDIM + col] =
          (_Float16)(ctx[dt][r] * inv);
    }
  }
}

// ---- residual + layernorm -------------------------------------------------

__global__ __launch_bounds__(256)
void ln_residual(const float* __restrict__ a, const float* __restrict__ b,
                 const float* __restrict__ g, const float* __restrict__ be,
                 float* __restrict__ outf, _Float16* __restrict__ outh) {
  const int row = blockIdx.x, tid = threadIdx.x;
  __shared__ float red[8];
  float v[4];
  float s = 0.f;
#pragma unroll
  for (int i = 0; i < 4; ++i) {
    int col = tid + i * 256;
    v[i] = a[(size_t)row * MDIM + col] + b[(size_t)row * MDIM + col];
    s += v[i];
  }
#pragma unroll
  for (int off = 16; off; off >>= 1) s += __shfl_xor(s, off, 32);
  if ((tid & 31) == 0) red[tid >> 5] = s;
  __syncthreads();
  float mean = 0.f;
#pragma unroll
  for (int j = 0; j < 8; ++j) mean += red[j];
  mean *= (1.f / MDIM);
  __syncthreads();
  float vs = 0.f;
#pragma unroll
  for (int i = 0; i < 4; ++i) { float d = v[i] - mean; vs += d * d; }
#pragma unroll
  for (int off = 16; off; off >>= 1) vs += __shfl_xor(vs, off, 32);
  if ((tid & 31) == 0) red[tid >> 5] = vs;
  __syncthreads();
  float var = 0.f;
#pragma unroll
  for (int j = 0; j < 8; ++j) var += red[j];
  float rstd = rsqrtf(var * (1.f / MDIM) + 1e-5f);
#pragma unroll
  for (int i = 0; i < 4; ++i) {
    int col = tid + i * 256;
    float o = (v[i] - mean) * rstd * g[col] + be[col];
    outf[(size_t)row * MDIM + col] = o;
    if (outh) outh[(size_t)row * MDIM + col] = (_Float16)o;
  }
}

// ---- launcher -------------------------------------------------------------

extern "C" void kernel_launch(void* const* d_in, const int* in_sizes, int n_in,
                              void* d_out, int out_size, void* d_ws, size_t ws_size,
                              hipStream_t stream) {
  const float* x     = (const float*)d_in[0];
  const int*   amask = (const int*)d_in[1];
  const float* Wq    = (const float*)d_in[2];
  const float* bq    = (const float*)d_in[3];
  const float* Wk    = (const float*)d_in[4];
  const float* bk    = (const float*)d_in[5];
  const float* Wv    = (const float*)d_in[6];
  const float* bv    = (const float*)d_in[7];
  const float* Wo    = (const float*)d_in[8];
  const float* bo    = (const float*)d_in[9];
  const float* g1    = (const float*)d_in[10];
  const float* b1    = (const float*)d_in[11];
  const float* W1    = (const float*)d_in[12];
  const float* bias1 = (const float*)d_in[13];
  const float* W2    = (const float*)d_in[14];
  const float* bias2 = (const float*)d_in[15];
  const float* g2    = (const float*)d_in[16];
  const float* b2    = (const float*)d_in[17];
  float* out = (float*)d_out;

  size_t off = 0;
  auto alloc = [&](size_t bytes) {
    size_t r = off;
    off = (off + bytes + 255) & ~(size_t)255;
    return (char*)d_ws + r;
  };
  _Float16* x_h    = (_Float16*)alloc((size_t)ROWS * MDIM * 2);        // 8 MB
  _Float16* wqkvT  = (_Float16*)alloc((size_t)3 * MDIM * MDIM * 2);    // 6 MB
  float*    bqkv   = (float*)   alloc((size_t)3 * MDIM * 4);
  _Float16* woT    = (_Float16*)alloc((size_t)MDIM * MDIM * 2);        // 2 MB
  _Float16* w1T    = (_Float16*)alloc((size_t)FFNH * MDIM * 2);        // 4 MB
  _Float16* w2T    = (_Float16*)alloc((size_t)MDIM * FFNH * 2);        // 4 MB
  _Float16* qk_h   = (_Float16*)alloc((size_t)ROWS * 2 * MDIM * 2);    // 16 MB
  _Float16* vT     = (_Float16*)alloc((size_t)BATCH * MDIM * SEQ * 2); // 8 MB
  _Float16* ctx_h  = (_Float16*)alloc((size_t)ROWS * MDIM * 2);        // 8 MB
  float*    attn_f = (float*)   alloc((size_t)ROWS * MDIM * 4);        // 16 MB
  float*    ln1_f  = (float*)   alloc((size_t)ROWS * MDIM * 4);        // 16 MB
  _Float16* ln1_h  = (_Float16*)alloc((size_t)ROWS * MDIM * 2);        // 8 MB
  _Float16* ffh_h  = qk_h;      // dead after attention; exactly 16 MB
  float*    ffn2_f = attn_f;    // dead after LN1

  const int T = 256;
  cast_f32_to_f16<<<(ROWS * MDIM + T - 1) / T, T, 0, stream>>>(x, x_h, ROWS * MDIM);
  pack_qkv_wT<<<(3 * MDIM * MDIM + T - 1) / T, T, 0, stream>>>(Wq, Wk, Wv, wqkvT);
  pack_qkv_b<<<(3 * MDIM + T - 1) / T, T, 0, stream>>>(bq, bk, bv, bqkv);
  tcast_f32_to_f16<<<(MDIM * MDIM + T - 1) / T, T, 0, stream>>>(Wo, woT, MDIM, MDIM);
  tcast_f32_to_f16<<<(MDIM * FFNH + T - 1) / T, T, 0, stream>>>(W1, w1T, MDIM, FFNH);
  tcast_f32_to_f16<<<(FFNH * MDIM + T - 1) / T, T, 0, stream>>>(W2, w2T, FFNH, MDIM);

  // QKV = x @ Wqkv + bqkv -> qk row-major + v transposed
  gemm_f16_wmma<2, false><<<dim3(3 * MDIM / 128, ROWS / 128), T, 0, stream>>>(
      x_h, wqkvT, bqkv, qk_h, vT, 3 * MDIM, MDIM);

  attn_flash<<<dim3(SEQ / 64, NHEAD, BATCH), 128, 0, stream>>>(qk_h, vT, amask, ctx_h);

  gemm_f16_wmma<0, false><<<dim3(MDIM / 128, ROWS / 128), T, 0, stream>>>(
      ctx_h, woT, bo, attn_f, nullptr, MDIM, MDIM);

  ln_residual<<<ROWS, T, 0, stream>>>(attn_f, x, g1, b1, ln1_f, ln1_h);

  gemm_f16_wmma<1, true><<<dim3(FFNH / 128, ROWS / 128), T, 0, stream>>>(
      ln1_h, w1T, bias1, ffh_h, nullptr, FFNH, MDIM);

  gemm_f16_wmma<0, false><<<dim3(MDIM / 128, ROWS / 128), T, 0, stream>>>(
      ffh_h, w2T, bias2, ffn2_f, nullptr, MDIM, FFNH);

  ln_residual<<<ROWS, T, 0, stream>>>(ln1_f, ffn2_f, g2, b2, out, nullptr);

  (void)in_sizes; (void)n_in; (void)out_size; (void)ws_size;
}
